// NonLocalBlock2D_12120397710080
// MI455X (gfx1250) — matrix operand order repaired and checked by reading, not compiled, not run on hardware
//
#include <hip/hip_runtime.h>
#include <hip/hip_bf16.h>

typedef __attribute__((ext_vector_type(16))) _Float16 v16h;
typedef __attribute__((ext_vector_type(8)))  _Float16 v8h;
typedef __attribute__((ext_vector_type(8)))  float    v8f;

#define C_IN  256
#define KD    32          // head dim / C_INT
#define NSEQ  4096        // H*W
#define NB    4           // batch

// ---------------------------------------------------------------------------
// Kernel 1: projections.  theta/phi/g[k,n] = sum_c w[k,c] * x[c,n]
// Outputs (f16):
//   Q    [B][N][32]  = theta^T  (row-major rows of Q -> WMMA A fragments)
//   PhiT [B][N][32]  = phi^T    (lane=column j, contiguous k -> WMMA B frag)
//   G    [B][32][N]  = g        (lane=column k, contiguous j -> WMMA B frag)
// ---------------------------------------------------------------------------
__global__ __launch_bounds__(256) void proj_kernel(
    const float* __restrict__ x,
    const float* __restrict__ w_theta,
    const float* __restrict__ w_phi,
    const float* __restrict__ w_g,
    _Float16* __restrict__ Q,
    _Float16* __restrict__ PhiT,
    _Float16* __restrict__ G)
{
    int tid = blockIdx.x * 256 + threadIdx.x;     // B*N*32 threads
    int k = tid & (KD - 1);
    int n = (tid >> 5) & (NSEQ - 1);
    int b = tid >> 17;                             // N*32 = 2^17

    const float* xb = x + ((size_t)b * C_IN) * NSEQ + n;
    float at = 0.f, ap = 0.f, ag = 0.f;
#pragma unroll 4
    for (int c = 0; c < C_IN; ++c) {
        float xv = xb[(size_t)c * NSEQ];           // uniform within wave
        at += w_theta[k * C_IN + c] * xv;
        ap += w_phi  [k * C_IN + c] * xv;
        ag += w_g    [k * C_IN + c] * xv;
    }
    size_t rn = ((size_t)b * NSEQ + n) * KD + k;
    Q[rn]    = (_Float16)at;
    PhiT[rn] = (_Float16)ap;
    G[((size_t)b * KD + k) * NSEQ + n] = (_Float16)ag;
}

// ---------------------------------------------------------------------------
// Kernel 2: fused attention (flash-style single pass, unnormalized softmax).
// One wave owns a 16-row Q tile; block = 4 waves = 64 rows; grid = B*N/64.
// Per 32-key chunk: S = Q x Phi (2 wmma), P = exp(S), O += P x G (2 wmma).
// ---------------------------------------------------------------------------
__global__ __launch_bounds__(128) void attn_kernel(
    const _Float16* __restrict__ Q,
    const _Float16* __restrict__ PhiT,
    const _Float16* __restrict__ G,
    float* __restrict__ Y)                         // [B][N][32] fp32
{
    __shared__ _Float16 ldsP[4][16 * 32];          // 1KB per wave P-staging

    const int lane = threadIdx.x & 31;
    const int wave = threadIdx.x >> 5;
    const int tilesPerB = NSEQ / 64;
    const int b    = blockIdx.x / tilesPerB;
    const int tile = blockIdx.x % tilesPerB;
    const int i0   = tile * 64 + wave * 16;        // query-row base of this wave

    const _Float16* Qb = Q    + (size_t)b * NSEQ * KD;
    const _Float16* Pb = PhiT + (size_t)b * NSEQ * KD;
    const _Float16* Gb = G    + (size_t)b * KD * NSEQ;

    const int r = lane & 15;                       // A: row / B: column
    const int h = lane >> 4;                       // half-wave selects K subset

    // ---- A fragment of Q (16x32 f16): element e -> K=16h+e
    v8h qlo = *(const v8h*)(Qb + (size_t)(i0 + r) * KD + 16 * h);
    v8h qhi = *(const v8h*)(Qb + (size_t)(i0 + r) * KD + 16 * h + 8);
    v16h aq = __builtin_shufflevector(qlo, qhi,
                0,1,2,3,4,5,6,7,8,9,10,11,12,13,14,15);

    v8f o0 = {}, o1 = {};                          // O tile: cols k 0..15 / 16..31
    float lsum[8];
#pragma unroll
    for (int v = 0; v < 8; ++v) lsum[v] = 0.f;

    _Float16* pts = &ldsP[wave][0];

    for (int jb = 0; jb < NSEQ; jb += 32) {
        // B fragments of phi (32x16): lane=col j, element e -> K = 16h+e
        v16h bp0 = *(const v16h*)(Pb + (size_t)(jb + r) * KD + 16 * h);
        v16h bp1 = *(const v16h*)(Pb + (size_t)(jb + 16 + r) * KD + 16 * h);

        v8f s0 = {}, s1 = {};
        s0 = __builtin_amdgcn_wmma_f32_16x16x32_f16(false, aq, false, bp0,
                                                    (short)0, s0, false, false);
        s1 = __builtin_amdgcn_wmma_f32_16x16x32_f16(false, aq, false, bp1,
                                                    (short)0, s1, false, false);

        // P = exp(S); accumulate per-row partial sums (row = v + 8h)
        float p0[8], p1[8];
#pragma unroll
        for (int v = 0; v < 8; ++v) {
            p0[v] = __expf(s0[v]);
            p1[v] = __expf(s1[v]);
            lsum[v] += p0[v] + p1[v];
        }

        // Stage P (16 rows x 32 j) in LDS row-major; C layout: row=v+8h, col=r
#pragma unroll
        for (int v = 0; v < 8; ++v) {
            pts[(v + 8 * h) * 32 + r]      = (_Float16)p0[v];
            pts[(v + 8 * h) * 32 + 16 + r] = (_Float16)p1[v];
        }
        // LDS ops are in-order within a wave (DScnt) — no barrier needed:
        // each wave reads only its own tile.

        // A fragment of P (16x32 over j)
        v8h plo = *(const v8h*)(pts + r * 32 + 16 * h);
        v8h phi = *(const v8h*)(pts + r * 32 + 16 * h + 8);
        v16h ap = __builtin_shufflevector(plo, phi,
                    0,1,2,3,4,5,6,7,8,9,10,11,12,13,14,15);

        // B fragments of g (32j x 16k): lane=col k, element e -> j = jb+16h+e
        v16h bg0 = *(const v16h*)(Gb + (size_t)r        * NSEQ + jb + 16 * h);
        v16h bg1 = *(const v16h*)(Gb + (size_t)(16 + r) * NSEQ + jb + 16 * h);

        o0 = __builtin_amdgcn_wmma_f32_16x16x32_f16(false, ap, false, bg0,
                                                    (short)0, o0, false, false);
        o1 = __builtin_amdgcn_wmma_f32_16x16x32_f16(false, ap, false, bg1,
                                                    (short)0, o1, false, false);
    }

    // Row sums: reduce across the 16 lanes of each half (rows v+8h)
#pragma unroll
    for (int v = 0; v < 8; ++v) {
        float s = lsum[v];
        s += __shfl_xor(s, 1, 32);
        s += __shfl_xor(s, 2, 32);
        s += __shfl_xor(s, 4, 32);
        s += __shfl_xor(s, 8, 32);
        lsum[v] = 1.0f / s;
    }

    float* Yb = Y + (size_t)b * NSEQ * KD;
#pragma unroll
    for (int v = 0; v < 8; ++v) {
        int row = i0 + v + 8 * h;                  // C/D layout row
        Yb[(size_t)row * KD + r]      = o0[v] * lsum[v];
        Yb[(size_t)row * KD + 16 + r] = o1[v] * lsum[v];
    }
}

// ---------------------------------------------------------------------------
// Kernel 3: out[b,c,n] = x[b,c,n] + sum_k w_out[c,k] * Y[b,n,k]
// ---------------------------------------------------------------------------
__global__ __launch_bounds__(256) void out_kernel(
    const float* __restrict__ x,
    const float* __restrict__ w_out,
    const float* __restrict__ Y,
    float* __restrict__ out)
{
    size_t idx = (size_t)blockIdx.x * 256 + threadIdx.x;  // B*C*N threads
    int n = (int)(idx & (NSEQ - 1));
    int c = (int)((idx >> 12) & (C_IN - 1));
    int b = (int)(idx >> 20);                              // C*N = 2^20

    const float* yrow = Y + ((size_t)b * NSEQ + n) * KD;
    const float* wrow = w_out + (size_t)c * KD;            // uniform per wave
    float acc = x[idx];
#pragma unroll
    for (int k = 0; k < KD; ++k) acc += wrow[k] * yrow[k];
    out[idx] = acc;
}

// ---------------------------------------------------------------------------
extern "C" void kernel_launch(void* const* d_in, const int* in_sizes, int n_in,
                              void* d_out, int out_size, void* d_ws, size_t ws_size,
                              hipStream_t stream)
{
    const float* x       = (const float*)d_in[0];
    const float* w_theta = (const float*)d_in[1];
    const float* w_phi   = (const float*)d_in[2];
    const float* w_g     = (const float*)d_in[3];
    const float* w_out   = (const float*)d_in[4];
    float* out = (float*)d_out;

    // Workspace: 3 x 1MB f16 + 2MB fp32 = 5MB
    _Float16* Q    = (_Float16*)d_ws;
    _Float16* PhiT = Q    + (size_t)NB * NSEQ * KD;
    _Float16* G    = PhiT + (size_t)NB * NSEQ * KD;
    float*    Y    = (float*)(G + (size_t)NB * KD * NSEQ);

    proj_kernel<<<(NB * NSEQ * KD) / 256, 256, 0, stream>>>(
        x, w_theta, w_phi, w_g, Q, PhiT, G);

    attn_kernel<<<NB * (NSEQ / 64), 128, 0, stream>>>(Q, PhiT, G, Y);

    out_kernel<<<(NB * C_IN * NSEQ) / 256, 256, 0, stream>>>(x, w_out, Y, out);
}
